// CombinedLoss_14362370638453
// MI455X (gfx1250) — compile-verified, matrix operand order checked by reading, not run on hardware
//
#include <hip/hip_runtime.h>
#include <stdint.h>

// ---------------------------------------------------------------------------
// CombinedLoss for MI455X (gfx1250, wave32, WMMA + TDM).
//   out[0]=total, out[1]=poisson, out[2]=contrastive
// bf16-normalize features once; three fused 4096x4096x1024 WMMA GEMMs that
// never materialize sim: each 256x128 block tile folds exp(sim/T) into
// per-row / per-col atomic accumulators (|sim|<=1 so exp needs no max
// subtraction) and stores the diagonal logits.  Operand tiles are DMA'd into
// LDS by the Tensor Data Mover (double-buffered, s_wait_tensorcnt), with TDM
// padding reproducing the 80B bank-spread row stride.  Each wave computes a
// 64x64 subtile (4x4 WMMA accums) -> 1.0 ds_load_b128 per WMMA.
// ---------------------------------------------------------------------------

typedef __attribute__((ext_vector_type(16))) __bf16       v16bf;
typedef __attribute__((ext_vector_type(8)))  float        v8f;
typedef __attribute__((ext_vector_type(4)))  unsigned int u32x4;
typedef __attribute__((ext_vector_type(8)))  int          i32x8;
typedef __attribute__((ext_vector_type(4)))  int          i32x4;

#define BSZ  4096
#define DSZ  1024
#define TM   256  // block tile rows (A side)
#define TN   128  // block tile cols (B side)
#define KT   32
#define LDSS 40   // shorts per LDS row: 32 data + 8 pad (80B stride, bank spread)

union Frag { v16bf v; uint4 q[2]; };

static __device__ __forceinline__ unsigned short f32_to_bf16_rne(float f) {
  union { float f; unsigned u; } cv; cv.f = f;
  unsigned u = cv.u;
  u += 0x7FFFu + ((u >> 16) & 1u);   // round-to-nearest-even (finite inputs)
  return (unsigned short)(u >> 16);
}

// --- TDM: DMA one rows x 32-col bf16 tile (row stride DSZ) into LDS. -------
// D# per ISA 8.3-8.6: 2D tensor, data_size=2B, pad 4 DWORDs after every
// 16 DWORDs (64B row -> 80B LDS stride, matching LDSS).
static __device__ __forceinline__ void tdm_load_tile(const unsigned short* gsrc,
                                                     unsigned lds_off, int rows) {
  const uint64_t ga = (uint64_t)(uintptr_t)gsrc;
  u32x4 g0;
  g0[0] = 1u;                                            // count=1 (valid D#)
  g0[1] = lds_off;                                       // lds_addr (bytes)
  g0[2] = (unsigned)ga;                                  // global_addr[31:0]
  g0[3] = ((unsigned)(ga >> 32) & 0x01FFFFFFu)           // global_addr[56:32]
          | (2u << 30);                                  // type=2 ("image")
  i32x8 g1;
  g1[0] = (1 << 16)      // data_size = 2 bytes
        | (1 << 20)      // pad_enable
        | (3 << 22)      // pad_interval: 16 DWORDs (one 64B row)
        | (3 << 25);     // pad_amount: 4 DWORDs (16B)
  g1[1] = (DSZ & 0xFFFF) << 16;                          // tensor_dim0[15:0]
  g1[2] = (DSZ >> 16) | ((BSZ & 0xFFFF) << 16);          // dim0[31:16]|dim1[15:0]
  g1[3] = (BSZ >> 16) | (KT << 16);                      // dim1[31:16]|tile_dim0
  g1[4] = rows;                                          // tile_dim1 (tile_dim2=0)
  g1[5] = DSZ;                                           // tensor_dim0_stride lo32
  g1[6] = 0;                                             // stride hi / dim1_stride
  g1[7] = 0;
  i32x4 g2 = {0, 0, 0, 0};                               // 2D: groups 2/3 unused
  i32x4 g3 = {0, 0, 0, 0};
#if defined(__clang_major__) && (__clang_major__ >= 23)
  i32x8 g4 = {0, 0, 0, 0, 0, 0, 0, 0};
  __builtin_amdgcn_tensor_load_to_lds(g0, g1, g2, g3, g4, 0);
#else
  __builtin_amdgcn_tensor_load_to_lds(g0, g1, g2, g3, 0);
#endif
}

// ---------------------------------------------------------------- zero scratch
__global__ void zero_acc_kernel(float* p, int n) {
  int i = blockIdx.x * blockDim.x + threadIdx.x;
  if (i < n) p[i] = 0.0f;
}

// ------------------------------------------------- row L2-normalize -> bf16
__global__ __launch_bounds__(256)
void normalize_rows_kernel(const float* __restrict__ f1,
                           const float* __restrict__ f2,
                           const float* __restrict__ f3,
                           unsigned short* __restrict__ out) {
  const int row  = blockIdx.x;
  const int feat = blockIdx.y;
  const int tid  = threadIdx.x;
  const float* src = (feat == 0 ? f1 : (feat == 1 ? f2 : f3)) + (size_t)row * DSZ;

  float ss = 0.0f;
  for (int i = tid; i < DSZ; i += 256) { float v = src[i]; ss += v * v; }
  for (int m = 1; m < 32; m <<= 1) ss += __shfl_xor(ss, m, 32);

  __shared__ float warpsum[8];
  __shared__ float total;
  if ((tid & 31) == 0) warpsum[tid >> 5] = ss;
  __syncthreads();
  if (tid == 0) {
    float t = 0.0f;
    for (int w = 0; w < 8; ++w) t += warpsum[w];
    total = t;
  }
  __syncthreads();

  const float inv = 1.0f / fmaxf(sqrtf(total), 1e-12f);
  unsigned short* dst = out + ((size_t)feat * BSZ + row) * DSZ;
  for (int i = tid; i < DSZ; i += 256) dst[i] = f32_to_bf16_rne(src[i] * inv);
}

// ------------------------------------------------------- Poisson partial sum
__global__ __launch_bounds__(256)
void poisson_kernel(const float* __restrict__ in, const float* __restrict__ tg,
                    float* __restrict__ acc, int n) {
  float s = 0.0f;
  for (int i = blockIdx.x * blockDim.x + threadIdx.x; i < n;
       i += gridDim.x * blockDim.x)
    s += in[i] - tg[i] * logf(in[i] + 1e-8f);
  for (int m = 1; m < 32; m <<= 1) s += __shfl_xor(s, m, 32);
  if ((threadIdx.x & 31) == 0) atomicAdd(acc, s);
}

// -------------------------------------- fused sim GEMM + online exp-sums
// grid = (16, 32, 3 pairs), block = 256 (8 waves) -> 256x128 tile.
// Wave w (wm = w>>1 in 0..3, wn = w&1) computes 64x64 = 4x4 WMMA tiles.
// TDM double-buffers operand tiles in LDS; one workgroup barrier per K-step.
__global__ __launch_bounds__(256)
void sim_gemm_kernel(const unsigned short* __restrict__ norm,
                     float* __restrict__ accbase) {
  const int pa   = blockIdx.z;               // pair: (0,1) (0,2) (1,2)
  const int aIdx = (pa == 2) ? 1 : 0;
  const int bIdx = (pa == 0) ? 1 : 2;
  const unsigned short* __restrict__ A  = norm + (size_t)aIdx * BSZ * DSZ;
  const unsigned short* __restrict__ Bm = norm + (size_t)bIdx * BSZ * DSZ;
  float* __restrict__ rowsum = accbase + (size_t)pa * BSZ;
  float* __restrict__ colsum = accbase + (size_t)(3 + pa) * BSZ;
  float* __restrict__ diag   = accbase + (size_t)(6 + pa) * BSZ;

  const int tileM = blockIdx.x, tileN = blockIdx.y;
  const int tid   = threadIdx.x;
  const int lane  = tid & 31;
  const int w     = tid >> 5;
  const int wm    = w >> 1;     // 0..3 -> 64-row slab of the 256-row tile
  const int wn    = w & 1;      // 0..1 -> 64-col slab of the 128-col tile
  const int lrow  = lane & 15;
  const int lhalf = lane >> 4;

  // [buffer][A rows 0..255 | B rows 0..127] * 80B rows: 2 x 30720B = 60KB
  __shared__ unsigned short lds[2][(TM + TN) * LDSS];

  const unsigned short* Abase = A  + (size_t)tileM * TM * DSZ;
  const unsigned short* Bbase = Bm + (size_t)tileN * TN * DSZ;

  const v8f vzero = {0.f, 0.f, 0.f, 0.f, 0.f, 0.f, 0.f, 0.f};
  v8f acc[4][4];
  #pragma unroll
  for (int mt = 0; mt < 4; ++mt)
    #pragma unroll
    for (int nt = 0; nt < 4; ++nt) acc[mt][nt] = vzero;

  // Prologue: DMA buffer 0 (wave 0 owns the TDM; EXEC is ignored by TDM ops)
  if (w == 0) {
    tdm_load_tile(Abase, (unsigned)(uintptr_t)&lds[0][0], TM);
    tdm_load_tile(Bbase, (unsigned)(uintptr_t)&lds[0][TM * LDSS], TN);
  }

  for (int k0 = 0; k0 < DSZ; k0 += KT) {
    const int cur = (k0 / KT) & 1;
    if (w == 0) __builtin_amdgcn_s_wait_tensorcnt(0);  // cur buffer landed
    __syncthreads();  // cur visible to all; all done reading buffer cur^1
    if (w == 0 && (k0 + KT) < DSZ) {                   // overlap DMA w/ WMMA
      tdm_load_tile(Abase + (k0 + KT),
                    (unsigned)(uintptr_t)&lds[cur ^ 1][0], TM);
      tdm_load_tile(Bbase + (k0 + KT),
                    (unsigned)(uintptr_t)&lds[cur ^ 1][TM * LDSS], TN);
    }

    const unsigned short* lA = &lds[cur][0];
    const unsigned short* lB = &lds[cur][TM * LDSS];

    // Load ALL fragments first so ds_loads pipeline, then 16 back-to-back
    // WMMAs.  Per-lane frag = 16 bf16 of one row with K split
    // {0..7,16..23}/{8..15,24..31} across lane halves (ISA 7.12.2).
    Frag fa[4], fb[4];
    #pragma unroll
    for (int nt = 0; nt < 4; ++nt) {
      const uint4* p = (const uint4*)&lB[(wn * 64 + nt * 16 + lrow) * LDSS];
      fb[nt].q[0] = p[lhalf];
      fb[nt].q[1] = p[2 + lhalf];
    }
    #pragma unroll
    for (int mt = 0; mt < 4; ++mt) {
      const uint4* p = (const uint4*)&lA[(wm * 64 + mt * 16 + lrow) * LDSS];
      fa[mt].q[0] = p[lhalf];
      fa[mt].q[1] = p[2 + lhalf];
    }
    #pragma unroll
    for (int mt = 0; mt < 4; ++mt)
      #pragma unroll
      for (int nt = 0; nt < 4; ++nt)
        acc[mt][nt] = __builtin_amdgcn_wmma_f32_16x16x32_bf16(
            false, fa[mt].v, false, fb[nt].v, (short)0, acc[mt][nt], false, false);
  }

  // ---- epilogue: logits l = sim/T (T=0.5); |sim|<=1 so exp is safe directly.
  const int m_base = tileM * TM + wm * 64;
  const int n_base = tileN * TN + wn * 64;
  float csum[4] = {0.0f, 0.0f, 0.0f, 0.0f};
  float rs[4][8];

  #pragma unroll
  for (int mt = 0; mt < 4; ++mt) {
    #pragma unroll
    for (int r = 0; r < 8; ++r) {
      const int gm = m_base + mt * 16 + r + lhalf * 8;  // C/D: M = r + 8*half
      float rsv = 0.0f;
      #pragma unroll
      for (int nt = 0; nt < 4; ++nt) {
        const float l  = acc[mt][nt][r] * 2.0f;
        const int   gn = n_base + nt * 16 + lrow;       // C/D: N = lane&15
        if (gm == gn) diag[gm] = l;                     // unique writer
        const float e = __expf(l);
        rsv += e;
        csum[nt] += e;
      }
      rs[mt][r] = rsv;
    }
  }
  // row partials: reduce across the 16 lanes holding one row's N values
  #pragma unroll
  for (int mt = 0; mt < 4; ++mt) {
    #pragma unroll
    for (int r = 0; r < 8; ++r) {
      float v = rs[mt][r];
      v += __shfl_xor(v, 1, 32);
      v += __shfl_xor(v, 2, 32);
      v += __shfl_xor(v, 4, 32);
      v += __shfl_xor(v, 8, 32);
      if (lrow == 0)
        atomicAdd(&rowsum[m_base + mt * 16 + r + lhalf * 8], v);
    }
  }
  // col partials: lanes L and L+16 hold the same N
  #pragma unroll
  for (int nt = 0; nt < 4; ++nt) {
    float c = csum[nt];
    c += __shfl_xor(c, 16, 32);
    if (lhalf == 0) atomicAdd(&colsum[n_base + nt * 16 + lrow], c);
  }
}

// --------------------------------------------------------------- finalize
__global__ __launch_bounds__(256)
void finalize_kernel(const float* __restrict__ accbase, float* __restrict__ out) {
  const int tid = threadIdx.x;
  float c = 0.0f;
  for (int i = tid; i < BSZ; i += 256) {
    #pragma unroll
    for (int p = 0; p < 3; ++p) {
      const float rsv = accbase[p * BSZ + i];
      const float csv = accbase[(3 + p) * BSZ + i];
      const float dv  = accbase[(6 + p) * BSZ + i];
      c += 0.5f * (logf(rsv) + logf(csv)) - dv;
    }
  }
  for (int m = 1; m < 32; m <<= 1) c += __shfl_xor(c, m, 32);
  __shared__ float wsum[8];
  if ((tid & 31) == 0) wsum[tid >> 5] = c;
  __syncthreads();
  if (tid == 0) {
    float ct = 0.0f;
    for (int w = 0; w < 8; ++w) ct += wsum[w];
    const float c_loss = ct / (float)BSZ / 3.0f;
    const float p_loss = accbase[9 * BSZ] / (float)((size_t)BSZ * DSZ);
    out[0] = p_loss + c_loss;
    out[1] = p_loss;
    out[2] = c_loss;
  }
}

// ---------------------------------------------------------------------------
extern "C" void kernel_launch(void* const* d_in, const int* in_sizes, int n_in,
                              void* d_out, int out_size, void* d_ws, size_t ws_size,
                              hipStream_t stream) {
  const float* inputs  = (const float*)d_in[0];
  const float* targets = (const float*)d_in[1];
  const float* f1      = (const float*)d_in[2];
  const float* f2      = (const float*)d_in[3];
  const float* f3      = (const float*)d_in[4];
  float* out = (float*)d_out;

  // workspace layout: [3 * 4096 * 1024 bf16 normalized features | 9*4096+1 f32]
  unsigned short* normw = (unsigned short*)d_ws;
  float* accbase = (float*)((char*)d_ws + (size_t)3 * BSZ * DSZ * sizeof(unsigned short));
  const int accN = 9 * BSZ + 1;  // rowsum[3]+colsum[3]+diag[3] per-row + poisson acc

  zero_acc_kernel<<<(accN + 255) / 256, 256, 0, stream>>>(accbase, accN);
  normalize_rows_kernel<<<dim3(BSZ, 3), 256, 0, stream>>>(f1, f2, f3, normw);
  poisson_kernel<<<2048, 256, 0, stream>>>(inputs, targets, accbase + 9 * BSZ,
                                           BSZ * DSZ);
  sim_gemm_kernel<<<dim3(BSZ / TM, BSZ / TN, 3), 256, 0, stream>>>(normw,
                                                                   accbase);
  finalize_kernel<<<1, 256, 0, stream>>>(accbase, out);
}
